// NeuronAudioAttention_32564442038534
// MI455X (gfx1250) — matrix operand order repaired and checked by reading, not compiled, not run on hardware
//
#include <hip/hip_runtime.h>
#include <cstdint>
#include <cstddef>

// ---------------------------------------------------------------------------
// Types for CDNA5 WMMA (wave32): 16x16x32 bf16 -> f32 accumulate
// ---------------------------------------------------------------------------
typedef __bf16 bf16_t;
typedef bf16_t v16bf __attribute__((ext_vector_type(16)));
typedef float  v8f   __attribute__((ext_vector_type(8)));

union Frag   { unsigned int u[8]; v16bf v; };
union Pack16 { uint4 q[2]; unsigned short s[16]; };

__device__ __forceinline__ unsigned short f2bf(float f) {
  // round-to-nearest-even fp32 -> bf16
  unsigned int u = __float_as_uint(f);
  u += 0x7FFFu + ((u >> 16) & 1u);
  return (unsigned short)(u >> 16);
}

// Low 32 bits of a generic pointer to LDS == byte offset in the LDS aperture.
__device__ __forceinline__ unsigned lds_off(const void* p) {
  return (unsigned)(unsigned long long)p;
}

// CDNA5 async DMA: two GLOBAL_LOAD_ASYNC_TO_LDS_B128 (32 contiguous bytes).
// IOFFSET is added to BOTH the global and the LDS address (ISA 15.18.3),
// so one address pair covers both 16B chunks. Tracked by ASYNCcnt (+2).
__device__ __forceinline__ void async_b128x2(unsigned lds, const void* g) {
  asm volatile(
      "global_load_async_to_lds_b128 %0, %1, off\n\t"
      "global_load_async_to_lds_b128 %0, %1, off offset:16"
      :: "v"(lds), "v"((unsigned long long)g) : "memory");
}

__device__ __forceinline__ void wait_async_le4() {
  asm volatile("s_wait_asynccnt 0x4" ::: "memory");
}
__device__ __forceinline__ void wait_async_le2() {
  asm volatile("s_wait_asynccnt 0x2" ::: "memory");
}
__device__ __forceinline__ void wait_async_0() {
  asm volatile("s_wait_asynccnt 0x0" ::: "memory");
}

// ---------------------------------------------------------------------------
// Kernel 1: fp32 -> bf16 elementwise (4 elems / thread)
// ---------------------------------------------------------------------------
__global__ void cvt_bf16_kernel(const float* __restrict__ in,
                                unsigned short* __restrict__ out, int n4) {
  int i = blockIdx.x * blockDim.x + threadIdx.x;
  if (i < n4) {
    float4 v = reinterpret_cast<const float4*>(in)[i];
    uint2 p;
    p.x = (unsigned)f2bf(v.x) | ((unsigned)f2bf(v.y) << 16);
    p.y = (unsigned)f2bf(v.z) | ((unsigned)f2bf(v.w) << 16);
    reinterpret_cast<uint2*>(out)[i] = p;
  }
}

// ---------------------------------------------------------------------------
// Kernel 2: Wt[n][k] = bf16(W[k][n])  (K,N multiples of 32)
// ---------------------------------------------------------------------------
__global__ __launch_bounds__(256)
void transpose_bf16_kernel(const float* __restrict__ W,
                           unsigned short* __restrict__ Wt, int K, int N) {
  __shared__ float tile[32][33];
  const int kx = blockIdx.x * 32, nx = blockIdx.y * 32;
  const int tx = threadIdx.x & 31, ty = threadIdx.x >> 5;
  #pragma unroll
  for (int r = ty; r < 32; r += 8)
    tile[r][tx] = W[(size_t)(kx + r) * N + (nx + tx)];
  __syncthreads();
  #pragma unroll
  for (int r = ty; r < 32; r += 8)
    Wt[(size_t)(nx + r) * K + (kx + tx)] = f2bf(tile[tx][r]);
}

// ---------------------------------------------------------------------------
// Kernel 3/5: C[M,N] = A[M,K] (bf16) @ Wt[N,K]^T (bf16) + bias
// Block tile 128x128, 8 waves, each wave: 16 rows x 128 cols
//   (8 x v_wmma_f32_16x16x32_bf16 per K-step of 32).
// Tiles DMAed into LDS with GLOBAL_LOAD_ASYNC_TO_LDS_B128, double-buffered.
// All 8 B-fragments are preloaded before the WMMA burst so the scheduler can
// overlap LDS reads with the matrix pipe instead of waiting per-WMMA.
// LDS row stride 56 halfwords = 112B: 16B-aligned rows for async B128 writes,
// and 28*r mod 64 covers all 16 banks for the fragment reads.
// Output: bf16 (Cbf) or fp32 (Cf).
// ---------------------------------------------------------------------------
__global__ __launch_bounds__(256, 2)
void wmma_gemm_kernel(const unsigned short* __restrict__ A,
                      const unsigned short* __restrict__ Bt,
                      const float* __restrict__ bias,
                      float* __restrict__ Cf,
                      unsigned short* __restrict__ Cbf,
                      int M, int N, int K) {
  __shared__ unsigned short As[2][128 * 56];
  __shared__ unsigned short Bs[2][128 * 56];

  const int t = threadIdx.x;
  const int wave = t >> 5;
  const int lane = t & 31;
  const int l16 = lane & 15;
  const int hi = (lane < 16) ? 0 : 1;
  const int mblk = blockIdx.x * 128;
  const int nblk = blockIdx.y * 128;
  const int kbaseA = hi ? 16 : 0;  // A k order used here (contiguous): hi lanes hold K 16..31
  const int kbaseB = hi ? 16 : 0;  // B k order used here (same as A): hi lanes hold K 16..31

  // Cooperative async tile issue: thread t -> tile row r=t>>1, 32B half-row.
  // Rows clamped in-range (never skipped) so per-wave ASYNCcnt stays uniform;
  // duplicated rows only feed output rows that are never stored.
  auto issue_tile = [&](unsigned short* lds, const unsigned short* src,
                        int row0, int rows_total, int kk) {
    const int r = t >> 1;
    const int half = t & 1;
    int grow = row0 + r;
    if (grow >= rows_total) grow = rows_total - 1;
    const unsigned short* g = src + (size_t)grow * K + kk + half * 16;
    async_b128x2(lds_off(lds + r * 56 + half * 16), g);
  };

  v8f acc[8];
  #pragma unroll
  for (int j = 0; j < 8; ++j) {
    float bv = bias ? bias[nblk + j * 16 + l16] : 0.0f;
    #pragma unroll
    for (int i = 0; i < 8; ++i) acc[j][i] = bv;
  }

  const int nk = K >> 5;  // K-steps of 32
  issue_tile(As[0], A, mblk, M, 0);
  issue_tile(Bs[0], Bt, nblk, N, 0);

  for (int ik = 0; ik < nk; ++ik) {
    const int cur = ik & 1;
    if (ik + 1 < nk) {
      // prefetch next tile into the alternate buffer, then wait for the
      // 4 older async loads only (async loads complete in order).
      issue_tile(As[cur ^ 1], A, mblk, M, (ik + 1) * 32);
      issue_tile(Bs[cur ^ 1], Bt, nblk, N, (ik + 1) * 32);
      wait_async_le4();
    } else {
      wait_async_0();
    }
    __syncthreads();  // current tiles resident for all waves

    Frag a;
    const unsigned short* ap = &As[cur][(wave * 16 + l16) * 56];
    #pragma unroll
    for (int d = 0; d < 8; ++d) {
      int K0 = kbaseA + 2 * d;
      a.u[d] = *reinterpret_cast<const unsigned int*>(ap + K0);
    }
    Frag bf[8];
    #pragma unroll
    for (int j = 0; j < 8; ++j) {
      const unsigned short* bp = &Bs[cur][(j * 16 + l16) * 56 + kbaseB];
      #pragma unroll
      for (int d = 0; d < 8; ++d)
        bf[j].u[d] = *reinterpret_cast<const unsigned int*>(bp + 2 * d);
    }
    #pragma unroll
    for (int j = 0; j < 8; ++j)
      acc[j] = __builtin_amdgcn_wmma_f32_16x16x32_bf16(
          false, a.v, false, bf[j].v, (short)0, acc[j], false, false);
    __syncthreads();  // all waves done reading before buffer is overwritten
  }

  // Epilogue: C layout VGPR i -> row i (lanes 0-15) / row 8+i (lanes 16-31)
  #pragma unroll
  for (int j = 0; j < 8; ++j) {
    const int col = nblk + j * 16 + l16;
    #pragma unroll
    for (int i = 0; i < 8; ++i) {
      const int row = mblk + wave * 16 + (hi ? 8 + i : i);
      if (row < M) {
        if (Cf) Cf[(size_t)row * N + col] = acc[j][i];
        else    Cbf[(size_t)row * N + col] = f2bf(acc[j][i]);
      }
    }
  }
}

// ---------------------------------------------------------------------------
// Kernel 4: flash attention, one block = (b,h) x 128 query rows, 8 waves.
// Software-pipelined over 64-key blocks with double-buffered LDS tiles:
//   K tile [64 keys][64 hd]  (stride 72 = 144B, async DMA, clamped)
//   V tile transposed [64 hd][64 keys] (register path: transpose + exact
//     zero-fill for OOB keys so P(=0) x V stays zero through the WMMA)
//   per-wave P scratch [16][64] (stride 72)
// Next block's K (async) and V (registers) are prefetched under the current
// block's 16 WMMAs + online softmax.
// ---------------------------------------------------------------------------
__global__ __launch_bounds__(256, 2)
void wmma_attn_kernel(const unsigned short* __restrict__ Q,
                      const unsigned short* __restrict__ Kmat,
                      const unsigned short* __restrict__ Vmat,
                      const float* __restrict__ mask,
                      unsigned short* __restrict__ Ctx,
                      int S, int Hn, int HDn, float scaling) {
  const int Dm = Hn * HDn;
  __shared__ unsigned short Ks[2][64 * 72];
  __shared__ unsigned short Vt[2][64 * 72];
  __shared__ unsigned short Ps[8][16 * 72];

  const int t = threadIdx.x;
  const int wave = t >> 5;
  const int lane = t & 31;
  const int l16 = lane & 15;
  const int hi = (lane < 16) ? 0 : 1;
  const int qblk = blockIdx.x * 128;
  const int b = blockIdx.y / Hn;
  const int h = blockIdx.y % Hn;
  const int kbaseA = hi ? 16 : 0;
  const int kbaseB = hi ? 16 : 0;

  // Q fragments (16 rows x 64 hd => two 16x32 A-fragments), loaded once.
  const int qrow = qblk + wave * 16 + l16;
  const int qr = (qrow < S) ? qrow : (S - 1);
  Frag qf[2];
  {
    const unsigned short* qp = Q + ((size_t)(b * S + qr)) * Dm + h * HDn;
    #pragma unroll
    for (int f = 0; f < 2; ++f)
      #pragma unroll
      for (int d = 0; d < 8; ++d) {
        int K0 = f * 32 + kbaseA + 2 * d;
        qf[f].u[d] = *reinterpret_cast<const unsigned int*>(qp + K0);
      }
  }

  float mrun[8], lrun[8];
  v8f o[4];
  #pragma unroll
  for (int i = 0; i < 8; ++i) { mrun[i] = -3.0e38f; lrun[i] = 0.0f; }
  #pragma unroll
  for (int tt = 0; tt < 4; ++tt)
    #pragma unroll
    for (int i = 0; i < 8; ++i) o[tt][i] = 0.0f;

  // Per-thread tile-load coordinates (thread -> key row, 16-elem hd segment).
  const int key_l = t >> 2;  // 0..63
  const int seg = t & 3;     // 0..3
  auto kv_base = [&](int kb) -> size_t {
    int key = kb * 64 + key_l;
    int keyc = (key < S) ? key : (S - 1);
    return ((size_t)(b * S + keyc)) * Dm + h * HDn + seg * 16;
  };
  auto load_v = [&](int kb, Pack16& v) {
    const int key = kb * 64 + key_l;
    #pragma unroll
    for (int e = 0; e < 16; ++e) v.s[e] = 0;
    if (key < S) {
      const uint4* vp = reinterpret_cast<const uint4*>(Vmat + kv_base(kb));
      v.q[0] = vp[0];
      v.q[1] = vp[1];
    }
  };

  const int nkb = (S + 63) >> 6;
  // Prologue: issue block 0's K async DMA, pull block 0's V into registers.
  async_b128x2(lds_off(&Ks[0][key_l * 72 + seg * 16]), Kmat + kv_base(0));
  Pack16 vreg;
  load_v(0, vreg);

  for (int kb = 0; kb < nkb; ++kb) {
    const int cur = kb & 1;
    // (a) commit this block's V registers into LDS (transposed).
    //     Safe: trailing barrier of iteration kb-1 guarantees Vt[cur] was
    //     fully consumed (last read in iteration kb-2).
    #pragma unroll
    for (int e = 0; e < 16; ++e)
      Vt[cur][(seg * 16 + e) * 72 + key_l] = vreg.s[e];
    // (b) prefetch next block under this block's compute.
    if (kb + 1 < nkb) {
      async_b128x2(lds_off(&Ks[cur ^ 1][key_l * 72 + seg * 16]),
                   Kmat + kv_base(kb + 1));
      load_v(kb + 1, vreg);
      wait_async_le2();  // the 2 older K loads (this block) have landed
    } else {
      wait_async_0();
    }
    __syncthreads();  // K async + V stores visible block-wide

    // scores = Q @ K^T : 4 column tiles of 16 keys
    v8f sc[4];
    #pragma unroll
    for (int j = 0; j < 4; ++j) {
      Frag kf0, kf1;
      const unsigned short* kp = &Ks[cur][(j * 16 + l16) * 72 + kbaseB];
      #pragma unroll
      for (int d = 0; d < 8; ++d)
        kf0.u[d] = *reinterpret_cast<const unsigned int*>(kp + 2 * d);
      #pragma unroll
      for (int d = 0; d < 8; ++d)
        kf1.u[d] = *reinterpret_cast<const unsigned int*>(kp + 32 + 2 * d);
      v8f z;
      #pragma unroll
      for (int i = 0; i < 8; ++i) z[i] = 0.0f;
      z = __builtin_amdgcn_wmma_f32_16x16x32_bf16(
          false, qf[0].v, false, kf0.v, (short)0, z, false, false);
      z = __builtin_amdgcn_wmma_f32_16x16x32_bf16(
          false, qf[1].v, false, kf1.v, (short)0, z, false, false);
      sc[j] = z;
    }

    // scale + attention mask + invalid-key masking
    #pragma unroll
    for (int j = 0; j < 4; ++j) {
      const int col = kb * 64 + j * 16 + l16;
      #pragma unroll
      for (int i = 0; i < 8; ++i) {
        const int row = qblk + wave * 16 + (hi ? 8 + i : i);
        float v = sc[j][i] * scaling;
        if (col < S) {
          if (row < S) v += mask[((size_t)(b * S + row)) * S + col];
        } else {
          v = -3.0e38f;
        }
        sc[j][i] = v;
      }
    }

    // online softmax (row stats live in the C-layout lane mapping)
    float corr[8], mnew[8];
    #pragma unroll
    for (int i = 0; i < 8; ++i) {
      float m = fmaxf(fmaxf(sc[0][i], sc[1][i]), fmaxf(sc[2][i], sc[3][i]));
      #pragma unroll
      for (int off = 1; off < 16; off <<= 1)
        m = fmaxf(m, __shfl_xor(m, off, 32));
      mnew[i] = fmaxf(mrun[i], m);
      corr[i] = __expf(mrun[i] - mnew[i]);
      mrun[i] = mnew[i];
    }
    #pragma unroll
    for (int i = 0; i < 8; ++i) {
      float ls = 0.0f;
      #pragma unroll
      for (int j = 0; j < 4; ++j) {
        float p = __expf(sc[j][i] - mnew[i]);
        sc[j][i] = p;
        ls += p;
      }
      #pragma unroll
      for (int off = 1; off < 16; off <<= 1) ls += __shfl_xor(ls, off, 32);
      lrun[i] = lrun[i] * corr[i] + ls;
    }
    #pragma unroll
    for (int tt = 0; tt < 4; ++tt)
      #pragma unroll
      for (int i = 0; i < 8; ++i) o[tt][i] *= corr[i];

    // relayout P: C-layout -> LDS -> A-fragments
    unsigned short* P = &Ps[wave][0];
    #pragma unroll
    for (int j = 0; j < 4; ++j)
      #pragma unroll
      for (int i = 0; i < 8; ++i) {
        const int prow = hi ? 8 + i : i;
        P[prow * 72 + j * 16 + l16] = f2bf(sc[j][i]);
      }
    __syncthreads();

    Frag pf[2];
    #pragma unroll
    for (int f = 0; f < 2; ++f)
      #pragma unroll
      for (int d = 0; d < 8; ++d) {
        int K0 = f * 32 + kbaseA + 2 * d;
        pf[f].u[d] = *reinterpret_cast<const unsigned int*>(P + l16 * 72 + K0);
      }

    // O += P @ V : 4 hd tiles of 16
    #pragma unroll
    for (int tt = 0; tt < 4; ++tt) {
      Frag vf0, vf1;
      const unsigned short* vp = &Vt[cur][(tt * 16 + l16) * 72 + kbaseB];
      #pragma unroll
      for (int d = 0; d < 8; ++d)
        vf0.u[d] = *reinterpret_cast<const unsigned int*>(vp + 2 * d);
      #pragma unroll
      for (int d = 0; d < 8; ++d)
        vf1.u[d] = *reinterpret_cast<const unsigned int*>(vp + 32 + 2 * d);
      o[tt] = __builtin_amdgcn_wmma_f32_16x16x32_bf16(
          false, pf[0].v, false, vf0.v, (short)0, o[tt], false, false);
      o[tt] = __builtin_amdgcn_wmma_f32_16x16x32_bf16(
          false, pf[1].v, false, vf1.v, (short)0, o[tt], false, false);
    }
    __syncthreads();  // block done with Ks[cur]/Vt[cur] before next overwrite
  }

  // normalize + store context (bf16, [B,S,H,HD] layout)
  #pragma unroll
  for (int tt = 0; tt < 4; ++tt) {
    const int hd = tt * 16 + l16;
    #pragma unroll
    for (int i = 0; i < 8; ++i) {
      const int row = qblk + wave * 16 + (hi ? 8 + i : i);
      if (row < S) {
        float denom = lrun[i];
        float val = o[tt][i] / (denom > 0.0f ? denom : 1.0f);
        Ctx[((size_t)(b * S + row)) * Dm + h * HDn + hd] = f2bf(val);
      }
    }
  }
}

// ---------------------------------------------------------------------------
// Host-side orchestration
// ---------------------------------------------------------------------------
extern "C" void kernel_launch(void* const* d_in, const int* in_sizes, int n_in,
                              void* d_out, int out_size, void* d_ws,
                              size_t ws_size, hipStream_t stream) {
  (void)in_sizes; (void)n_in; (void)out_size; (void)ws_size;

  const int B = 4, S = 1500, D = 1280, H = 20, HD = 64;
  const int M = B * S;  // 6000 tokens
  const float scaling = 0.125f;  // HD^-0.5 = 64^-0.5

  const float* hidden = (const float*)d_in[0];
  const float* mask   = (const float*)d_in[1];
  const float* q_w = (const float*)d_in[2];
  const float* q_b = (const float*)d_in[3];
  const float* k_w = (const float*)d_in[4];
  const float* v_w = (const float*)d_in[5];
  const float* v_b = (const float*)d_in[6];
  const float* o_w = (const float*)d_in[7];
  const float* o_b = (const float*)d_in[8];
  float* out = (float*)d_out;

  char* ws = (char*)d_ws;
  size_t off = 0;
  auto alloc = [&](size_t bytes) -> void* {
    void* p = ws + off;
    off += (bytes + 255) & ~(size_t)255;
    return p;
  };
  unsigned short* Xbf = (unsigned short*)alloc((size_t)M * D * 2);
  unsigned short* Wtq = (unsigned short*)alloc((size_t)D * D * 2);
  unsigned short* Wtk = (unsigned short*)alloc((size_t)D * D * 2);
  unsigned short* Wtv = (unsigned short*)alloc((size_t)D * D * 2);
  unsigned short* Wto = (unsigned short*)alloc((size_t)D * D * 2);
  unsigned short* Qbf = (unsigned short*)alloc((size_t)M * D * 2);
  unsigned short* Kbf = (unsigned short*)alloc((size_t)M * D * 2);
  unsigned short* Vbf = (unsigned short*)alloc((size_t)M * D * 2);
  unsigned short* Ctx = (unsigned short*)alloc((size_t)M * D * 2);

  // 1) hidden -> bf16
  {
    int n4 = (M * D) / 4;
    cvt_bf16_kernel<<<(n4 + 255) / 256, 256, 0, stream>>>(hidden, Xbf, n4);
  }
  // 2) transposed bf16 weights
  {
    dim3 g(D / 32, D / 32);
    transpose_bf16_kernel<<<g, 256, 0, stream>>>(q_w, Wtq, D, D);
    transpose_bf16_kernel<<<g, 256, 0, stream>>>(k_w, Wtk, D, D);
    transpose_bf16_kernel<<<g, 256, 0, stream>>>(v_w, Wtv, D, D);
    transpose_bf16_kernel<<<g, 256, 0, stream>>>(o_w, Wto, D, D);
  }
  // 3) Q/K/V projections (bf16 out; k_proj has no bias)
  {
    dim3 g((M + 127) / 128, D / 128);
    wmma_gemm_kernel<<<g, 256, 0, stream>>>(Xbf, Wtq, q_b, nullptr, Qbf, M, D, D);
    wmma_gemm_kernel<<<g, 256, 0, stream>>>(Xbf, Wtk, nullptr, nullptr, Kbf, M, D, D);
    wmma_gemm_kernel<<<g, 256, 0, stream>>>(Xbf, Wtv, v_b, nullptr, Vbf, M, D, D);
  }
  // 4) flash attention
  {
    dim3 g((S + 127) / 128, B * H);
    wmma_attn_kernel<<<g, 256, 0, stream>>>(Qbf, Kbf, Vbf, mask, Ctx, S, H, HD,
                                            scaling);
  }
  // 5) output projection (fp32 out, with bias)
  {
    dim3 g((M + 127) / 128, D / 128);
    wmma_gemm_kernel<<<g, 256, 0, stream>>>(Ctx, Wto, o_b, out, nullptr, M, D, D);
  }
}
